// DecoderRNN_72559177498917
// MI455X (gfx1250) — matrix operand order repaired and checked by reading, not compiled, not run on hardware
//
#include <hip/hip_runtime.h>

// Sizes from the reference
#define BB 32
#define TT 64
#define VV 32000
#define EE 256
#define HH 512
#define GG 2048   // 4*H
#define KX 768    // H + E

typedef _Float16 v16h __attribute__((ext_vector_type(16)));
typedef _Float16 v8h  __attribute__((ext_vector_type(8)));
typedef float    v8f  __attribute__((ext_vector_type(8)));

// ---- WMMA fragment helpers (contiguous k = 16h + s order, same in A and B) -----------------

// A fragment: 16x32 f16. lane l: row = l&15, h = l>>4.
// elements 0..15 -> K = kb + h*16 + 0..15
__device__ __forceinline__ v16h ldA(const _Float16* rowbase, int kb, int h) {
  const _Float16* p = rowbase + kb + h * 16;
  v8h lo = *(const v8h*)(p);
  v8h hi = *(const v8h*)(p + 8);
  return __builtin_shufflevector(lo, hi, 0,1,2,3,4,5,6,7,8,9,10,11,12,13,14,15);
}

// B fragment: 32x16 f16, B(k,n) = W[n][k] (row-major [N,K] weights).
// lane l: col = l&15, K = kb + (l>>4)*16 + 0..15 (contiguous -> 32B load)
__device__ __forceinline__ v16h ldB(const _Float16* W, int n, int K, int kb, int h) {
  return *(const v16h*)(W + (size_t)n * K + kb + h * 16);
}

__device__ __forceinline__ v8f wmma_f16(v16h a, v16h b, v8f c) {
  return __builtin_amdgcn_wmma_f32_16x16x32_f16(false, a, false, b, (short)0, c,
                                                false, false);
}

__device__ __forceinline__ float sigf(float x) {
  return 1.0f / (1.0f + __expf(-x));
}
__device__ __forceinline__ float tanh_fast(float x) {
  x = fminf(fmaxf(x, -15.0f), 15.0f);
  float e = __expf(2.0f * x);
  return (e - 1.0f) / (e + 1.0f);
}

// ---- prologue kernels ------------------------------------------------------

__global__ void cvt_f16_kernel(const float* __restrict__ src,
                               _Float16* __restrict__ dst, int n) {
  int i = blockIdx.x * 256 + threadIdx.x;
  if (i < n) dst[i] = (_Float16)src[i];
}

__global__ void bias_sum_kernel(const float* __restrict__ a,
                                const float* __restrict__ b,
                                float* __restrict__ d, int n) {
  int i = blockIdx.x * 256 + threadIdx.x;
  if (i < n) d[i] = a[i] + b[i];
}

// e_seq[t][b][k]: t==0 -> embed_table[0], else embed_table[captions[b][t-1]]
__global__ void embed_kernel(const int* __restrict__ cap,
                             const float* __restrict__ table,
                             _Float16* __restrict__ ebf) {
  int idx = blockIdx.x * 256 + threadIdx.x;   // < 64*32*256 = 2^19
  int k = idx & (EE - 1);
  int b = (idx >> 8) & (BB - 1);
  int t = idx >> 13;
  int row = (t == 0) ? 0 : cap[b * TT + (t - 1)];
  ebf[idx] = (_Float16)table[(size_t)row * EE + k];
}

// hS0 = features @ sz_w.T + sz_b ; zero the other initial states
__global__ void init_state_kernel(const float* __restrict__ feat,
                                  const float* __restrict__ szw,
                                  const float* __restrict__ szb,
                                  float* __restrict__ hS0,
                                  float* __restrict__ cS0,
                                  float* __restrict__ hU0,
                                  float* __restrict__ cU0) {
  int idx = blockIdx.x * 256 + threadIdx.x;   // < 32*512
  if (idx >= BB * HH) return;
  int b = idx >> 9, j = idx & (HH - 1);
  const float* f = feat + (size_t)b * EE;
  const float* w = szw + (size_t)j * EE;
  float s = szb[j];
  for (int k = 0; k < EE; ++k) s += f[k] * w[k];
  hS0[idx] = s;
  cS0[idx] = 0.0f;
  hU0[idx] = 0.0f;
  cU0[idx] = 0.0f;
}

// ---- fused LSTM cell (one timestep, one cell) ------------------------------
// gates = [xa(512) | e(256)] @ Wx.T + hprev @ Wh.T + bsum
// block bx owns hidden columns [bx*32, bx*32+32) of all 4 gates.
// 8 waves: wave w -> gate g=w>>1, 16-col half (w&1); each wave 2 M-tiles.
__global__ __launch_bounds__(256, 4) void lstm_cell_kernel(
    const float* __restrict__ xa, const _Float16* __restrict__ et,
    const float* __restrict__ hprev, const float* __restrict__ cprev,
    const _Float16* __restrict__ Wx, const _Float16* __restrict__ Wh,
    const float* __restrict__ bsum,
    float* __restrict__ hout, float* __restrict__ cout,
    _Float16* __restrict__ souts) {
  __shared__ _Float16 astage[BB * KX];   // 48 KB: x, then reused for h
  __shared__ float gsm[BB * 128];        // 16 KB gate staging

  const int tid = threadIdx.x;
  const int lane = tid & 31;
  const int wave = tid >> 5;
  const int h = lane >> 4;
  const int nl = lane & 15;
  const int jg = blockIdx.x * 32;                       // hidden col base
  const int g = wave >> 1;                              // gate index
  const int colbase = g * HH + jg + (wave & 1) * 16;    // row of Wx/Wh

  // stage x = [u_prev | e_t] as f16 [32][768]
  for (int idx = tid; idx < BB * KX; idx += 256) {
    int b = idx / KX, k = idx - b * KX;
    astage[idx] = (k < HH) ? (_Float16)xa[b * HH + k] : et[b * EE + (k - HH)];
  }
  __syncthreads();

  float bias = bsum[colbase + nl];
  v8f acc0 = {}, acc1 = {};
  for (int i = 0; i < 8; ++i) { acc0[i] = bias; acc1[i] = bias; }

  const _Float16* ar0 = astage + (size_t)nl * KX;         // rows 0..15
  const _Float16* ar1 = astage + (size_t)(16 + nl) * KX;  // rows 16..31
#pragma unroll 4
  for (int kb = 0; kb < KX; kb += 32) {
    v16h bf = ldB(Wx, colbase + nl, KX, kb, h);
    v16h a0 = ldA(ar0, kb, h);
    v16h a1 = ldA(ar1, kb, h);
    acc0 = wmma_f16(a0, bf, acc0);
    acc1 = wmma_f16(a1, bf, acc1);
  }
  __syncthreads();

  // re-stage h_prev as f16 [32][512]
  for (int idx = tid; idx < BB * HH; idx += 256)
    astage[idx] = (_Float16)hprev[idx];
  __syncthreads();

  const _Float16* hr0 = astage + (size_t)nl * HH;
  const _Float16* hr1 = astage + (size_t)(16 + nl) * HH;
#pragma unroll 4
  for (int kb = 0; kb < HH; kb += 32) {
    v16h bf = ldB(Wh, colbase + nl, HH, kb, h);
    v16h a0 = ldA(hr0, kb, h);
    v16h a1 = ldA(hr1, kb, h);
    acc0 = wmma_f16(a0, bf, acc0);
    acc1 = wmma_f16(a1, bf, acc1);
  }

  // scatter gate tiles to LDS: gsm[row][g*32 + (wave&1)*16 + nl]
  int gcol = g * 32 + (wave & 1) * 16 + nl;
#pragma unroll
  for (int i = 0; i < 8; ++i) {
    gsm[(h * 8 + i) * 128 + gcol] = acc0[i];
    gsm[(16 + h * 8 + i) * 128 + gcol] = acc1[i];
  }
  __syncthreads();

  // fused activations + state update for hidden cols jg..jg+31
  for (int e = tid; e < BB * 32; e += 256) {
    int b = e >> 5, jl = e & 31;
    float iv = gsm[b * 128 + jl];
    float fv = gsm[b * 128 + 32 + jl];
    float gv = gsm[b * 128 + 64 + jl];
    float ov = gsm[b * 128 + 96 + jl];
    int o = b * HH + jg + jl;
    float cn = sigf(fv) * cprev[o] + sigf(iv) * tanh_fast(gv);
    float hn = sigf(ov) * tanh_fast(cn);
    cout[o] = cn;
    hout[o] = hn;
    if (souts) souts[o] = (_Float16)hn;   // f16 copy feeds the FC GEMM
  }
}

// ---- final FC: logits = souts[0..62] @ fc_w.T + fc_b -----------------------
// A: [2016][512] f16 rows (t*32+b); grid (125 N-blocks x 63 M-blocks)
// block: 32 rows x 256 cols; A tile staged once in LDS (shared by 8 waves);
// wave w -> cols [w*32, w*32+32), 4 accumulators (2 M-tiles x 2 N-tiles)
__global__ __launch_bounds__(256, 4) void fc_kernel(
    const _Float16* __restrict__ A, const _Float16* __restrict__ W,
    const float* __restrict__ bias, float* __restrict__ out) {
  __shared__ _Float16 as[BB * HH];   // 32 KB
  const int tid = threadIdx.x;
  const int lane = tid & 31;
  const int wave = tid >> 5;
  const int h = lane >> 4;
  const int nl = lane & 15;
  const int nbase = blockIdx.x * 256 + wave * 32;
  const int mbase = blockIdx.y * 32;
  const int n0 = nbase + nl, n1 = nbase + 16 + nl;

  // cooperative A staging: 32 rows x 512 cols, vectorized 8-half chunks
  {
    const v8h* src = (const v8h*)(A + (size_t)mbase * HH);
    v8h* dst = (v8h*)as;
    for (int idx = tid; idx < (BB * HH) / 8; idx += 256) dst[idx] = src[idx];
  }
  __syncthreads();

  float b0 = bias[n0], b1 = bias[n1];
  v8f acc00 = {}, acc01 = {}, acc10 = {}, acc11 = {};
#pragma unroll
  for (int i = 0; i < 8; ++i) {
    acc00[i] = b0; acc01[i] = b1; acc10[i] = b0; acc11[i] = b1;
  }

  const _Float16* ar0 = as + (size_t)nl * HH;
  const _Float16* ar1 = as + (size_t)(16 + nl) * HH;
#pragma unroll 4
  for (int kb = 0; kb < HH; kb += 32) {
    v16h a0 = ldA(ar0, kb, h);
    v16h a1 = ldA(ar1, kb, h);
    v16h bf0 = ldB(W, n0, HH, kb, h);
    v16h bf1 = ldB(W, n1, HH, kb, h);
    acc00 = wmma_f16(a0, bf0, acc00);
    acc01 = wmma_f16(a0, bf1, acc01);
    acc10 = wmma_f16(a1, bf0, acc10);
    acc11 = wmma_f16(a1, bf1, acc11);
  }

  // out[b][t][v], row gm = t*32 + b
#pragma unroll
  for (int i = 0; i < 8; ++i) {
    int gm0 = mbase + h * 8 + i;
    int gm1 = gm0 + 16;
    int t0 = gm0 >> 5, bb0 = gm0 & 31;
    int t1 = gm1 >> 5, bb1 = gm1 & 31;
    size_t r0 = ((size_t)bb0 * (TT - 1) + t0) * VV;
    size_t r1 = ((size_t)bb1 * (TT - 1) + t1) * VV;
    out[r0 + n0] = acc00[i];
    out[r0 + n1] = acc01[i];
    out[r1 + n0] = acc10[i];
    out[r1 + n1] = acc11[i];
  }
}

// ---- host ------------------------------------------------------------------

extern "C" void kernel_launch(void* const* d_in, const int* in_sizes, int n_in,
                              void* d_out, int out_size, void* d_ws, size_t ws_size,
                              hipStream_t stream) {
  const float* features = (const float*)d_in[0];
  const int*   captions = (const int*)d_in[1];
  const float* embed    = (const float*)d_in[2];
  const float* WihS     = (const float*)d_in[3];
  const float* WhhS     = (const float*)d_in[4];
  const float* bihS     = (const float*)d_in[5];
  const float* bhhS     = (const float*)d_in[6];
  const float* WihU     = (const float*)d_in[7];
  const float* WhhU     = (const float*)d_in[8];
  const float* bihU     = (const float*)d_in[9];
  const float* bhhU     = (const float*)d_in[10];
  const float* fc_w     = (const float*)d_in[11];
  const float* fc_b     = (const float*)d_in[12];
  const float* sz_w     = (const float*)d_in[13];
  const float* sz_b     = (const float*)d_in[14];
  float* out = (float*)d_out;

  char* ws = (char*)d_ws;
  size_t off = 0;
  auto alloc = [&](size_t bytes) -> char* {
    char* p = ws + off;
    off += (bytes + 255) & ~(size_t)255;
    return p;
  };

  _Float16* WxS  = (_Float16*)alloc((size_t)GG * KX * 2);
  _Float16* WhS  = (_Float16*)alloc((size_t)GG * HH * 2);
  _Float16* WxU  = (_Float16*)alloc((size_t)GG * KX * 2);
  _Float16* WhU  = (_Float16*)alloc((size_t)GG * HH * 2);
  _Float16* fcw  = (_Float16*)alloc((size_t)VV * HH * 2);
  float*    bsS  = (float*)alloc((size_t)GG * 4);
  float*    bsU  = (float*)alloc((size_t)GG * 4);
  _Float16* ebf  = (_Float16*)alloc((size_t)TT * BB * EE * 2);
  float*    hS0b = (float*)alloc((size_t)BB * HH * 4);
  float*    hS1b = (float*)alloc((size_t)BB * HH * 4);
  float*    cS0b = (float*)alloc((size_t)BB * HH * 4);
  float*    cS1b = (float*)alloc((size_t)BB * HH * 4);
  float*    hU0b = (float*)alloc((size_t)BB * HH * 4);
  float*    hU1b = (float*)alloc((size_t)BB * HH * 4);
  float*    cU0b = (float*)alloc((size_t)BB * HH * 4);
  float*    cU1b = (float*)alloc((size_t)BB * HH * 4);
  _Float16* souts = (_Float16*)alloc((size_t)TT * BB * HH * 2);

  float* hS[2] = {hS0b, hS1b};
  float* cS[2] = {cS0b, cS1b};
  float* hU[2] = {hU0b, hU1b};
  float* cU[2] = {cU0b, cU1b};

  // 1) weight conversion f32 -> f16
  auto cvt = [&](const float* s, _Float16* d, int n) {
    cvt_f16_kernel<<<(n + 255) / 256, 256, 0, stream>>>(s, d, n);
  };
  cvt(WihS, WxS, GG * KX);
  cvt(WhhS, WhS, GG * HH);
  cvt(WihU, WxU, GG * KX);
  cvt(WhhU, WhU, GG * HH);
  cvt(fc_w, fcw, VV * HH);

  // 2) bias pre-sums
  bias_sum_kernel<<<(GG + 255) / 256, 256, 0, stream>>>(bihS, bhhS, bsS, GG);
  bias_sum_kernel<<<(GG + 255) / 256, 256, 0, stream>>>(bihU, bhhU, bsU, GG);

  // 3) embedding gather (start token + teacher forcing), f16
  embed_kernel<<<(TT * BB * EE) / 256, 256, 0, stream>>>(captions, embed, ebf);

  // 4) initial states
  init_state_kernel<<<(BB * HH + 255) / 256, 256, 0, stream>>>(
      features, sz_w, sz_b, hS[0], cS[0], hU[0], cU[0]);

  // 5) sequential LSTM scan (only t=0..62 contribute to logits)
  for (int t = 0; t < TT - 1; ++t) {
    int cur = t & 1, nxt = cur ^ 1;
    const _Float16* et = ebf + (size_t)t * BB * EE;
    // S-LSTM: x = [u_prev | e_t]
    lstm_cell_kernel<<<16, 256, 0, stream>>>(
        hU[cur], et, hS[cur], cS[cur], WxS, WhS, bsS,
        hS[nxt], cS[nxt], souts + (size_t)t * BB * HH);
    // U-LSTM: x = [hS_new | e_t]
    lstm_cell_kernel<<<16, 256, 0, stream>>>(
        hS[nxt], et, hU[cur], cU[cur], WxU, WhU, bsU,
        hU[nxt], cU[nxt], (_Float16*)nullptr);
  }

  // 6) big FC: [2016,512] @ [512,32000] + b  (dominant 66 GFLOP GEMM)
  fc_kernel<<<dim3(VV / 256, (TT - 1) * BB / 32), 256, 0, stream>>>(
      souts, fcw, fc_b, out);

  (void)in_sizes; (void)n_in; (void)out_size; (void)ws_size;
}
